// QuantizedLinear_90589450207298
// MI455X (gfx1250) — matrix operand order repaired and checked by reading, not compiled, not run on hardware
//
#include <hip/hip_runtime.h>
#include <hip/hip_bf16.h>
#include <cstdint>

// ---------------------------------------------------------------------------
// out[M=4096, N=16384] = X[4096,4096] @ W[16384,4096]^T + bias
// W[o,i] = code[q[o,i]] * absmax[o]   (BLOCK == in_features == 4096)
// Strategy: dequant W -> bf16 (128 MB, L2-resident), X -> bf16 (32 MB),
// then double-buffered async-copy WMMA GEMM (bf16 in, f32 accumulate).
// ---------------------------------------------------------------------------
#define GEMM_M 4096
#define GEMM_N 16384
#define GEMM_K 4096

typedef __attribute__((ext_vector_type(16))) __bf16 v16bf;
typedef __attribute__((ext_vector_type(8)))  __bf16 v8bf;
typedef __attribute__((ext_vector_type(8)))  float  v8f;
typedef __attribute__((ext_vector_type(4)))  int    v4i;

#if defined(__has_builtin)
#if __has_builtin(__builtin_amdgcn_global_load_async_to_lds_b128)
#define HAVE_ASYNC_LDS 1
#endif
#if __has_builtin(__builtin_amdgcn_s_wait_asynccnt)
#define HAVE_WAIT_ASYNC 1
#endif
#endif

static __device__ __forceinline__ __bf16 f32_to_bf16_rne(float f) {
    union { float f; uint32_t u; } in; in.f = f;
    uint32_t r = in.u + 0x7FFFu + ((in.u >> 16) & 1u);   // round to nearest even
    union { unsigned short s; __bf16 b; } out; out.s = (unsigned short)(r >> 16);
    return out.b;
}

// 16-byte global -> LDS copy; async (no VGPR round-trip) when available.
// Builtin signature (probe-confirmed from clang diagnostic):
//   (v4i addrspace(1)* gsrc, v4i addrspace(3)* ldst, Imm int offset, Imm int cpol)
static __device__ __forceinline__ void copy16_g2l(const __bf16* g, __bf16* l) {
#if defined(HAVE_ASYNC_LDS)
    __builtin_amdgcn_global_load_async_to_lds_b128(
        (__attribute__((address_space(1))) v4i*)g,
        (__attribute__((address_space(3))) v4i*)l, 0, 0);
#else
    *reinterpret_cast<uint4*>(l) = *reinterpret_cast<const uint4*>(g);
#endif
}

static __device__ __forceinline__ void wait_async_copies() {
#if defined(HAVE_ASYNC_LDS)
#if defined(HAVE_WAIT_ASYNC)
    __builtin_amdgcn_s_wait_asynccnt(0);
#else
    asm volatile("s_wait_asynccnt 0x0" ::: "memory");
#endif
#endif
}

// ---------------------------------------------------------------------------
// Kernel 1: dequantize W (int8 codes stored as int32) -> bf16, 8 elems/thread.
// ---------------------------------------------------------------------------
__global__ void dequant_w_bf16(const int* __restrict__ q,
                               const float* __restrict__ absmax,
                               const float* __restrict__ code,
                               __bf16* __restrict__ w) {
    __shared__ float lut[256];
    lut[threadIdx.x] = code[threadIdx.x];
    __syncthreads();

    const long base = ((long)blockIdx.x * 256 + threadIdx.x) * 8;
    const int  row  = (int)(base >> 12);            // 4096 elems per row
    const float am  = absmax[row];

    const int4* qv = reinterpret_cast<const int4*>(q + base);
    int4 c0 = qv[0];
    int4 c1 = qv[1];

    union { v8bf v; __bf16 e[8]; } o;
    o.e[0] = f32_to_bf16_rne(lut[c0.x & 255] * am);
    o.e[1] = f32_to_bf16_rne(lut[c0.y & 255] * am);
    o.e[2] = f32_to_bf16_rne(lut[c0.z & 255] * am);
    o.e[3] = f32_to_bf16_rne(lut[c0.w & 255] * am);
    o.e[4] = f32_to_bf16_rne(lut[c1.x & 255] * am);
    o.e[5] = f32_to_bf16_rne(lut[c1.y & 255] * am);
    o.e[6] = f32_to_bf16_rne(lut[c1.z & 255] * am);
    o.e[7] = f32_to_bf16_rne(lut[c1.w & 255] * am);
    *reinterpret_cast<v8bf*>(w + base) = o.v;
}

// ---------------------------------------------------------------------------
// Kernel 2: convert X f32 -> bf16, 8 elems / thread.
// ---------------------------------------------------------------------------
__global__ void convert_x_bf16(const float* __restrict__ x,
                               __bf16* __restrict__ xb) {
    const long base = ((long)blockIdx.x * 256 + threadIdx.x) * 8;
    const float4* xv = reinterpret_cast<const float4*>(x + base);
    float4 a = xv[0];
    float4 b = xv[1];
    union { v8bf v; __bf16 e[8]; } o;
    o.e[0] = f32_to_bf16_rne(a.x); o.e[1] = f32_to_bf16_rne(a.y);
    o.e[2] = f32_to_bf16_rne(a.z); o.e[3] = f32_to_bf16_rne(a.w);
    o.e[4] = f32_to_bf16_rne(b.x); o.e[5] = f32_to_bf16_rne(b.y);
    o.e[6] = f32_to_bf16_rne(b.z); o.e[7] = f32_to_bf16_rne(b.w);
    *reinterpret_cast<v8bf*>(xb + base) = o.v;
}

// ---------------------------------------------------------------------------
// Kernel 3: bf16 WMMA GEMM, double-buffered LDS with async global->LDS copies.
// Block: 256 threads (8 waves). Block tile 128(M) x 128(N), K-step 64.
// Wave grid 2(M) x 4(N); each wave owns 64x32 = 4x2 wmma tiles.
// ---------------------------------------------------------------------------
#define BM 128
#define BN 128
#define BK 64
#define LDT (BK + 8)              // padded LDS row stride; 144 B, 16B-aligned
#define TILE_ELEMS ((BM + BN) * LDT)

__global__ void __launch_bounds__(256)
gemm_bf16_wmma(const __bf16* __restrict__ A,    // [M, K] activations (bf16)
               const __bf16* __restrict__ B,    // [N, K] dequant weights (bf16)
               const float*  __restrict__ bias, // [N]
               float* __restrict__ C) {         // [M, N]
    __shared__ __align__(16) __bf16 smem[2 * TILE_ELEMS];

    const int tid    = threadIdx.x;
    const int bn0    = blockIdx.x * BN;
    const int bm0    = blockIdx.y * BM;
    const int wave   = tid >> 5;
    const int lane   = tid & 31;
    const int l16    = lane & 15;
    const int hi     = lane >> 4;
    const int wave_m = (wave & 1) * 64;    // 0 or 64 within block tile
    const int wave_n = (wave >> 1) * 32;   // 0,32,64,96 within block tile

    v8f acc[4][2] = {};

    // stage K-tile `kk` into buffer `buf` (A: 1024 chunks, B: 1024 chunks;
    // 8 x 16B async copies per thread)
    auto stage = [&](int kk, int buf) {
        __bf16* sA = smem + buf * TILE_ELEMS;
        __bf16* sB = sA + BM * LDT;
        #pragma unroll
        for (int i = 0; i < 4; ++i) {
            int c   = tid + i * 256;
            int row = c >> 3;               // BK/8 = 8 chunks per row
            int col = (c & 7) * 8;
            copy16_g2l(&A[(long)(bm0 + row) * GEMM_K + kk + col], &sA[row * LDT + col]);
        }
        #pragma unroll
        for (int i = 0; i < 4; ++i) {
            int c   = tid + i * 256;
            int row = c >> 3;
            int col = (c & 7) * 8;
            copy16_g2l(&B[(long)(bn0 + row) * GEMM_K + kk + col], &sB[row * LDT + col]);
        }
    };

    // prologue: fill buffer 0
    stage(0, 0);
    wait_async_copies();
    __syncthreads();

    const int KT = GEMM_K / BK;            // 64 K-tiles
    for (int kt = 0; kt < KT; ++kt) {
        const int cur = kt & 1;
        // overlap: start async fill of the other buffer while we compute
        if (kt + 1 < KT) stage((kt + 1) * BK, cur ^ 1);

        const __bf16* sA = smem + cur * TILE_ELEMS;
        const __bf16* sB = sA + BM * LDT;

        #pragma unroll
        for (int ks = 0; ks < BK; ks += 32) {
            // A fragment (ISA 7.12.2, 16-bit A 16x32): lane (m=l16, hi):
            //   VGPR0-3 K = hi*8..hi*8+7 ; VGPR4-7 K = 16+hi*8..16+hi*8+7
            v16bf afrag[4];
            #pragma unroll
            for (int mi = 0; mi < 4; ++mi) {
                const __bf16* base = &sA[(wave_m + mi * 16 + l16) * LDT + ks];
                union { v16bf v; v8bf h[2]; } u;
                u.h[0] = *reinterpret_cast<const v8bf*>(base + hi * 8);
                u.h[1] = *reinterpret_cast<const v8bf*>(base + 16 + hi * 8);
                afrag[mi] = u.v;
            }
            // B fragment (16-bit B 32x16): lane n=l16 holds
            // K = hi*8..hi*8+7 then 16+hi*8..16+hi*8+7 (same as A).
            v16bf bfrag[2];
            #pragma unroll
            for (int ni = 0; ni < 2; ++ni) {
                const __bf16* base = &sB[(wave_n + ni * 16 + l16) * LDT + ks];
                union { v16bf v; v8bf h[2]; } u;
                u.h[0] = *reinterpret_cast<const v8bf*>(base + hi * 8);
                u.h[1] = *reinterpret_cast<const v8bf*>(base + 16 + hi * 8);
                bfrag[ni] = u.v;
            }
            #pragma unroll
            for (int mi = 0; mi < 4; ++mi)
                #pragma unroll
                for (int ni = 0; ni < 2; ++ni)
                    acc[mi][ni] = __builtin_amdgcn_wmma_f32_16x16x32_bf16(
                        false, afrag[mi], false, bfrag[ni],
                        (short)0, acc[mi][ni], false, false);
        }

        // next buffer must be fully written and every wave done reading `cur`
        wait_async_copies();
        __syncthreads();
    }

    // epilogue: C/D layout: VGPR r -> M = r + 8*hi, lane -> N = l16
    #pragma unroll
    for (int ni = 0; ni < 2; ++ni) {
        const int gn = bn0 + wave_n + ni * 16 + l16;
        const float bv = bias[gn];
        #pragma unroll
        for (int mi = 0; mi < 4; ++mi) {
            const int gm0 = bm0 + wave_m + mi * 16 + hi * 8;
            #pragma unroll
            for (int r = 0; r < 8; ++r)
                C[(long)(gm0 + r) * GEMM_N + gn] = acc[mi][ni][r] + bv;
        }
    }
}

// ---------------------------------------------------------------------------
// Launch: d_in = {x, qweight, absmax, code, bias}
// Workspace: [0, 128MB) = W bf16 ; [128MB, 160MB) = X bf16
// ---------------------------------------------------------------------------
extern "C" void kernel_launch(void* const* d_in, const int* in_sizes, int n_in,
                              void* d_out, int out_size, void* d_ws, size_t ws_size,
                              hipStream_t stream) {
    (void)in_sizes; (void)n_in; (void)out_size; (void)ws_size;

    const float* x      = (const float*)d_in[0];
    const int*   qw     = (const int*)  d_in[1];
    const float* absmax = (const float*)d_in[2];
    const float* code   = (const float*)d_in[3];
    const float* bias   = (const float*)d_in[4];
    float*       out    = (float*)d_out;

    __bf16* wb = (__bf16*)d_ws;                                        // 128 MB
    __bf16* xb = (__bf16*)((char*)d_ws + (size_t)GEMM_N * GEMM_K * 2); // +32 MB

    dequant_w_bf16<<<32768, 256, 0, stream>>>(qw, absmax, code, wb);
    convert_x_bf16<<<8192, 256, 0, stream>>>(x, xb);

    dim3 grid(GEMM_N / BN, GEMM_M / BM);   // (128, 32)
    gemm_bf16_wmma<<<grid, 256, 0, stream>>>(xb, wb, bias, out);
}
